// LSTMbyHand_74629351735854
// MI455X (gfx1250) — matrix operand order repaired and checked by reading, not compiled, not run on hardware
//
#include <hip/hip_runtime.h>

// MI455X / gfx1250 LSTM forward.
//   B=16, S=256, V=32000, H=512, L=2
// Strategy:
//  - all GEMMs via v_wmma_f32_16x16x32_bf16 (bf16 in, f32 accum)
//  - weights pre-transposed to bf16 [N][K] so B-fragments are contiguous loads
//  - recurrent part: one persistent 1024-thread (32-wave) workgroup per layer,
//    h/c state in LDS, wave w owns N-tile w (4 gates, 128 WMMA per step);
//    x_{t+1} panel prefetched (global_prefetch_b8) during elementwise phase
//  - logits: [4096,512]x[512,32000] GEMM; each wave computes 8 M-tiles x 1
//    N-tile so every B-fragment is reused by 8 WMMAs; B rows prefetched
//    4 k-steps ahead

#define B_ 16
#define S_ 256
#define V_ 32000
#define H_ 512
#define NTILES_V (V_ / 16)   // 2000
#define MT_ 8                // M-tiles (timesteps) per wave in logits GEMM

typedef __attribute__((ext_vector_type(16))) __bf16 v16bf;
typedef __attribute__((ext_vector_type(8)))  __bf16 v8bf;
typedef __attribute__((ext_vector_type(8)))  float  v8f;

// ---------------------------------------------------------------------------
// A-fragment (16xK tile, M=16 rows) in contiguous k order (B uses the same):
//   lane L: row = L&15; elements e=0..15 -> K = k0 + 16*(L>>4) + e
// Source is row-major with leading dim ld (K contiguous) -> two 16B loads.
// ---------------------------------------------------------------------------
__device__ __forceinline__ v16bf load_a_frag(const __bf16* __restrict__ base,
                                             int ld, int lane, int k0) {
  const int row = lane & 15;
  const int kb  = k0 + 16 * (lane >> 4);
  const v8bf lo = *(const v8bf*)(base + (size_t)row * ld + kb);
  const v8bf hi = *(const v8bf*)(base + (size_t)row * ld + kb + 8);
  v16bf a;
#pragma unroll
  for (int i = 0; i < 8; ++i) { a[i] = lo[i]; a[i + 8] = hi[i]; }
  return a;
}

// B-fragment (Kx16 tile), same contiguous k order as A: lane L holds column N = L&15,
// K = k0 + 16*(L>>4) + e, e=0..15 contiguous. Weights are pre-transposed to
// [N][K] row-major -> one 32B contiguous load per lane.
__device__ __forceinline__ v16bf load_b_frag(const __bf16* __restrict__ wT,
                                             int ld, int lane, int ntile,
                                             int k0) {
  const int n  = (ntile << 4) + (lane & 15);
  const int kb = k0 + 16 * (lane >> 4);
  return *(const v16bf*)(wT + (size_t)n * ld + kb);
}

// ---------------------------------------------------------------------------
// Tiled f32 -> bf16 transpose: dst[c][r] = src[r][c]; rows, cols % 32 == 0
// ---------------------------------------------------------------------------
__global__ void __launch_bounds__(256)
transpose_f32_to_bf16(__bf16* __restrict__ dst, const float* __restrict__ src,
                      int rows, int cols) {
  __shared__ float tile[32][33];
  const int c0 = blockIdx.x * 32;
  const int r0 = blockIdx.y * 32;
  const int tx = threadIdx.x;
  const int ty = threadIdx.y;
#pragma unroll
  for (int i = 0; i < 32; i += 8)
    tile[ty + i][tx] = src[(size_t)(r0 + ty + i) * cols + (c0 + tx)];
  __syncthreads();
#pragma unroll
  for (int i = 0; i < 32; i += 8)
    dst[(size_t)(c0 + ty + i) * rows + (r0 + tx)] = (__bf16)tile[tx][ty + i];
}

// ---------------------------------------------------------------------------
// Embedding gather -> time-major bf16 activations x[t][b][h]
// ---------------------------------------------------------------------------
__global__ void __launch_bounds__(128)
gather_embed(const int* __restrict__ seq, const float* __restrict__ emb,
             __bf16* __restrict__ xout) {
  const int r = blockIdx.x;          // r = t*16 + b
  const int t = r >> 4, b = r & 15;
  const int tok = seq[b * S_ + t];
  const float* src = emb + (size_t)tok * H_;
  __bf16* dst = xout + (size_t)r * H_;
  for (int j = threadIdx.x; j < H_; j += 128) dst[j] = (__bf16)src[j];
}

// ---------------------------------------------------------------------------
// One LSTM layer, persistent single workgroup (32 waves = 32 N-tiles).
// Wt layout: 8 matrices [W_f,W_i,W_c,W_o,U_f,U_i,U_c,U_o], each bf16 [N][K].
// xin/hout: time-major bf16 [S][B][H].  h,c state in LDS.
// ---------------------------------------------------------------------------
__global__ void __launch_bounds__(1024)
lstm_layer(const __bf16* __restrict__ xin, const __bf16* __restrict__ Wt,
           const float* __restrict__ bf_, const float* __restrict__ bi_,
           const float* __restrict__ bc_, const float* __restrict__ bo_,
           __bf16* __restrict__ hout) {
  __shared__ __bf16 h_sh[B_][H_];   // 16 KB
  __shared__ float  c_sh[B_][H_];   // 32 KB

  for (int i = threadIdx.x; i < B_ * H_; i += 1024) {
    (&h_sh[0][0])[i] = (__bf16)0.0f;
    (&c_sh[0][0])[i] = 0.0f;
  }
  __syncthreads();

  const int wave = threadIdx.x >> 5;   // 0..31 == owned N-tile
  const int lane = threadIdx.x & 31;
  const int row  = lane & 15;
  const int half = lane >> 4;
  const int n    = (wave << 4) + row;  // global output column this lane owns

  const float bias0 = bf_[n];
  const float bias1 = bi_[n];
  const float bias2 = bc_[n];
  const float bias3 = bo_[n];

  for (int t = 0; t < S_; ++t) {
    const __bf16* xt = xin + (size_t)t * (B_ * H_);
    v8f acc[4] = {v8f{}, v8f{}, v8f{}, v8f{}};

#pragma unroll 2
    for (int kt = 0; kt < H_ / 32; ++kt) {
      const int k0 = kt * 32;
      // A frag from x_t (global) and from h_{t-1} (LDS)
      v16bf ax = load_a_frag(xt, H_, lane, k0);
      v16bf ah;
      {
        const int kb = k0 + 16 * half;
        const v8bf lo = *(const v8bf*)&h_sh[row][kb];
        const v8bf hi = *(const v8bf*)&h_sh[row][kb + 8];
#pragma unroll
        for (int i = 0; i < 8; ++i) { ah[i] = lo[i]; ah[i + 8] = hi[i]; }
      }
#pragma unroll
      for (int g = 0; g < 4; ++g) {
        v16bf bw = load_b_frag(Wt + (size_t)g * H_ * H_,       H_, lane, wave, k0);
        v16bf bu = load_b_frag(Wt + (size_t)(g + 4) * H_ * H_, H_, lane, wave, k0);
        acc[g] = __builtin_amdgcn_wmma_f32_16x16x32_bf16(
            false, ax, false, bw, (short)0, acc[g], false, false);
        acc[g] = __builtin_amdgcn_wmma_f32_16x16x32_bf16(
            false, ah, false, bu, (short)0, acc[g], false, false);
      }
    }

    __syncthreads();  // everyone done reading h_{t-1}

    // Cooperative prefetch of next timestep's x panel (16 KB: 1024 thr x 8
    // bf16) -> global_prefetch_b8; uniform branch, no WMMA divergence.
    if (t + 1 < S_)
      __builtin_prefetch(xin + (size_t)(t + 1) * (B_ * H_) +
                             (size_t)threadIdx.x * 8, 0, 3);

    // Elementwise gate math on the 8 accumulator rows this lane owns:
    // C/D layout: vgpr r, lanes 0-15 -> M=r, lanes 16-31 -> M=r+8
#pragma unroll
    for (int r = 0; r < 8; ++r) {
      const int bidx = r + 8 * half;
      const float fp = acc[0][r] + bias0;
      const float ip = acc[1][r] + bias1;
      const float gp = acc[2][r] + bias2;
      const float op = acc[3][r] + bias3;
      const float f = 1.0f / (1.0f + __expf(-fp));
      const float i = 1.0f / (1.0f + __expf(-ip));
      const float g = tanhf(gp);
      const float o = 1.0f / (1.0f + __expf(-op));
      const float cnew = f * c_sh[bidx][n] + i * g;
      c_sh[bidx][n] = cnew;
      const float hv = o * tanhf(cnew);
      const __bf16 hb = (__bf16)hv;
      h_sh[bidx][n] = hb;
      hout[(size_t)t * (B_ * H_) + (size_t)bidx * H_ + n] = hb;
    }

    __syncthreads();  // h_t visible before next step's reads
  }
}

// ---------------------------------------------------------------------------
// Logits GEMM: [S*B, H] x [H, V] + b_out.
// Each wave: MT_=8 M-tiles (8 consecutive timesteps) x 1 N-tile; every
// B-fragment load feeds 8 WMMAs. All 8 waves of a block share one t-group,
// so A-fragments hit L1. h2 time-major [t][b][h]; WoT bf16 [V][H];
// out f32 [B][S][V].
// ---------------------------------------------------------------------------
__global__ void __launch_bounds__(256)
logits_gemm(const __bf16* __restrict__ h2, const __bf16* __restrict__ WoT,
            const float* __restrict__ bout, float* __restrict__ out) {
  const int wave = threadIdx.x >> 5;
  const int lane = threadIdx.x & 31;
  const int job  = blockIdx.x * 8 + wave;  // 0 .. (S_/MT_)*NTILES_V - 1
  const int tg   = job / NTILES_V;         // t-group: timesteps tg*MT_ ..
  const int nt   = job % NTILES_V;
  const __bf16* Abase = h2 + (size_t)tg * MT_ * (B_ * H_);

  const int half = lane >> 4;
  const int n    = (nt << 4) + (lane & 15);
  const __bf16* Brow = WoT + (size_t)n * H_ + 16 * half;  // this lane's B row

  v8f acc[MT_];
#pragma unroll
  for (int m = 0; m < MT_; ++m) acc[m] = v8f{};

#pragma unroll 2
  for (int kt = 0; kt < H_ / 32; ++kt) {
    const int k0 = kt * 32;
    const v16bf b = *(const v16bf*)(Brow + k0);
    // Stay 4 k-steps (128 elements) ahead of the streaming B walk.
    if (kt + 4 < H_ / 32) __builtin_prefetch(Brow + k0 + 128, 0, 0);
#pragma unroll
    for (int m = 0; m < MT_; ++m) {
      v16bf a = load_a_frag(Abase + (size_t)m * (B_ * H_), H_, lane, k0);
      acc[m] = __builtin_amdgcn_wmma_f32_16x16x32_bf16(
          false, a, false, b, (short)0, acc[m], false, false);
    }
  }

  const float bias = bout[n];
#pragma unroll
  for (int m = 0; m < MT_; ++m) {
    const int t = tg * MT_ + m;
#pragma unroll
    for (int r = 0; r < 8; ++r) {
      const int b_idx = r + 8 * half;
      out[(size_t)b_idx * ((size_t)S_ * V_) + (size_t)t * V_ + n] =
          acc[m][r] + bias;
    }
  }
}

// ---------------------------------------------------------------------------
extern "C" void kernel_launch(void* const* d_in, const int* in_sizes, int n_in,
                              void* d_out, int out_size, void* d_ws,
                              size_t ws_size, hipStream_t stream) {
  (void)in_sizes; (void)n_in; (void)out_size; (void)ws_size;

  const int*   seq   = (const int*)d_in[0];
  const float* emb   = (const float*)d_in[1];
  const float* W_f   = (const float*)d_in[2];
  const float* U_f   = (const float*)d_in[3];
  const float* b_f   = (const float*)d_in[4];
  const float* W_i   = (const float*)d_in[5];
  const float* U_i   = (const float*)d_in[6];
  const float* b_i   = (const float*)d_in[7];
  const float* W_c   = (const float*)d_in[8];
  const float* U_c   = (const float*)d_in[9];
  const float* b_c   = (const float*)d_in[10];
  const float* W_o   = (const float*)d_in[11];
  const float* U_o   = (const float*)d_in[12];
  const float* b_o   = (const float*)d_in[13];
  const float* W_out = (const float*)d_in[14];
  const float* b_out = (const float*)d_in[15];
  float* out = (float*)d_out;

  // Workspace layout (all bf16):
  //   WT     : 16 x [512][512] transposed recurrent weights   (8 MB)
  //   WoT    : [V][H] transposed output weights               (32.75 MB)
  //   xb     : [S][B][H] embedded inputs                      (4 MB)
  //   h1, h2 : [S][B][H] layer outputs                        (4 MB each)
  char* ws = (char*)d_ws;
  __bf16* WT  = (__bf16*)ws;
  __bf16* WoT = (__bf16*)(ws + (size_t)16 * H_ * H_ * 2);
  __bf16* xb  = (__bf16*)(ws + (size_t)16 * H_ * H_ * 2 + (size_t)V_ * H_ * 2);
  __bf16* h1  = xb + (size_t)S_ * B_ * H_;
  __bf16* h2  = h1 + (size_t)S_ * B_ * H_;

  const dim3 tb(32, 8);
  for (int l = 0; l < 2; ++l) {
    const float* srcs[8] = {
        W_f + (size_t)l * H_ * H_, W_i + (size_t)l * H_ * H_,
        W_c + (size_t)l * H_ * H_, W_o + (size_t)l * H_ * H_,
        U_f + (size_t)l * H_ * H_, U_i + (size_t)l * H_ * H_,
        U_c + (size_t)l * H_ * H_, U_o + (size_t)l * H_ * H_};
    for (int m = 0; m < 8; ++m)
      transpose_f32_to_bf16<<<dim3(H_ / 32, H_ / 32), tb, 0, stream>>>(
          WT + (size_t)(l * 8 + m) * H_ * H_, srcs[m], H_, H_);
  }
  transpose_f32_to_bf16<<<dim3(V_ / 32, H_ / 32), tb, 0, stream>>>(
      WoT, W_out, H_, V_);

  gather_embed<<<S_ * B_, 128, 0, stream>>>(seq, emb, xb);

  lstm_layer<<<1, 1024, 0, stream>>>(xb, WT, b_f, b_i, b_c, b_o, h1);
  lstm_layer<<<1, 1024, 0, stream>>>(h1, WT + (size_t)8 * H_ * H_,
                                     b_f + H_, b_i + H_, b_c + H_, b_o + H_, h2);

  logits_gemm<<<((S_ / MT_) * NTILES_V) / 8, 256, 0, stream>>>(
      h2, WoT, b_out, out);
}
